// RelationalAttention_73847667687682
// MI455X (gfx1250) — matrix operand order repaired and checked by reading, not compiled, not run on hardware
//
#include <hip/hip_runtime.h>
#include <hip/hip_bf16.h>
#include <math.h>

// ---------------------------------------------------------------------------
// RelationalAttention for MI455X (gfx1250, wave32, WMMA 16x16x32 f16)
//   B=2, N=1024, E=256, H=8, D=32, BIAS_HIDDEN=64
//   k0: f32->f16 weight packs (stored W^T)  k3: QKV GEMM (WMMA) + fused RoPE
//   k1: LayerNorm -> h (f16)                k4: flash attn + shared rel-bias MLP
//   k2: per-token bias-MLP precompute       k5: out-proj GEMM (WMMA) + bias
// All WMMA operand chunks are 16B-contiguous (b128); A and B tile staging both
// use GLOBAL_LOAD_ASYNC_TO_LDS_B128 (ASYNCcnt) — weights are pre-transposed so
// the LDS B^T tile is row-contiguous against global memory.
// ---------------------------------------------------------------------------

typedef __attribute__((ext_vector_type(16))) _Float16 v16h;
typedef __attribute__((ext_vector_type(8)))  float    v8f;
typedef __attribute__((__vector_size__(16))) int      v4i;

__device__ __forceinline__ v8f wmma_f16(v16h a, v16h b, v8f c) {
  return __builtin_amdgcn_wmma_f32_16x16x32_f16(
      false, a, false, b, (short)0, c, false, false);
}

// Build a 16xf16 fragment from two 16B-aligned 8-half chunks (b128 loads).
__device__ __forceinline__ v16h frag16(const _Float16* p0, const _Float16* p1) {
  union { uint4 u[2]; v16h v; } f;
  f.u[0] = *(const uint4*)p0;
  f.u[1] = *(const uint4*)p1;
  return f.v;
}

// ---- CDNA5 async global->LDS staging (ASYNCcnt), with portable fallback ----
#if __has_builtin(__builtin_amdgcn_global_load_async_to_lds_b128) && \
    __has_builtin(__builtin_amdgcn_s_wait_asynccnt)
#define USE_ASYNC_LDS 1
typedef __attribute__((address_space(1))) v4i g_v4i;   // global int4
typedef __attribute__((address_space(3))) v4i l_v4i;   // LDS int4
#else
#define USE_ASYNC_LDS 0
#endif

__device__ __forceinline__ void async_copy16(const _Float16* g, _Float16* l) {
#if USE_ASYNC_LDS
  // generic addr low 32 bits == LDS offset (aperture rule); global flat == VA
  __builtin_amdgcn_global_load_async_to_lds_b128(
      (g_v4i*)(uintptr_t)g, (l_v4i*)(uintptr_t)l, 0, 0);
#else
  *(uint4*)l = *(const uint4*)g;
#endif
}
__device__ __forceinline__ void async_wait0() {
#if USE_ASYNC_LDS
  __builtin_amdgcn_s_wait_asynccnt(0);
#endif
}

#define EMBED 256
#define NTOK  2048
#define NSEQ  1024
#define HEADS 8
#define HDIM  32
#define BH    64

// ------------------------------------- k0: weights -> f16, stored TRANSPOSED
// wqkv_t: [col 768][k 256], wo_t: [col 256][k 256]
__global__ void convert_weights_kernel(const float* __restrict__ wqkv,
                                       const float* __restrict__ wo,
                                       _Float16* __restrict__ wqkv_t,
                                       _Float16* __restrict__ wo_t) {
  int i = blockIdx.x * 256 + threadIdx.x;
  for (int idx = i; idx < EMBED * 3 * EMBED; idx += gridDim.x * 256) {
    int k = idx / 768, c = idx - k * 768;
    wqkv_t[c * EMBED + k] = (_Float16)wqkv[idx];
  }
  for (int idx = i; idx < EMBED * EMBED; idx += gridDim.x * 256) {
    int k = idx >> 8, c = idx & 255;
    wo_t[c * EMBED + k] = (_Float16)wo[idx];
  }
}

// -------------------------------------------------------------- k1: layernorm
__global__ void layernorm_kernel(const float* __restrict__ x,
                                 const float* __restrict__ g,
                                 const float* __restrict__ be,
                                 _Float16* __restrict__ out) {
  int row  = blockIdx.x * 8 + (threadIdx.x >> 5);
  int lane = threadIdx.x & 31;
  const float* xr = x + (size_t)row * EMBED;
  float v[8], s = 0.f, sq = 0.f;
#pragma unroll
  for (int i = 0; i < 8; ++i) {
    v[i] = xr[lane + 32 * i];
    s += v[i]; sq += v[i] * v[i];
  }
#pragma unroll
  for (int off = 16; off >= 1; off >>= 1) {
    s  += __shfl_xor(s,  off, 32);
    sq += __shfl_xor(sq, off, 32);
  }
  float mu  = s * (1.f / EMBED);
  float var = sq * (1.f / EMBED) - mu * mu;
  float inv = rsqrtf(var + 1e-5f);
#pragma unroll
  for (int i = 0; i < 8; ++i) {
    int c = lane + 32 * i;
    out[(size_t)row * EMBED + c] = (_Float16)((v[i] - mu) * inv * g[c] + be[c]);
  }
}

// ------------------------------------- k2: separable part of bias-MLP layer 1
__global__ void token_feat_kernel(const float* __restrict__ coords,
                                  const float* __restrict__ vel,
                                  const float* __restrict__ w1,
                                  _Float16* __restrict__ Ph) {
  int idx = blockIdx.x * 256 + threadIdx.x;
  int t = idx >> 6, c = idx & 63;
  float c0 = coords[t * 2], c1 = coords[t * 2 + 1];
  float v0 = vel[t * 2],    v1 = vel[t * 2 + 1];
  float p = c0 * w1[c] + c1 * w1[64 + c] + v0 * w1[192 + c] + v1 * w1[256 + c];
  Ph[idx] = (_Float16)p;
}

// ------------------------------------------------ k3: QKV GEMM + RoPE epilogue
// M=2048, N=768, K=256.  Block tile 64x64 (8 waves in 4x2, each 16x32).
// wBt is W^T [col][k] -> both A and B tiles staged fully async.
__global__ void __launch_bounds__(256)
qkv_gemm_rope_kernel(const _Float16* __restrict__ hA,
                     const _Float16* __restrict__ wBt,
                     _Float16* __restrict__ qh, _Float16* __restrict__ kh,
                     _Float16* __restrict__ vh) {
  __shared__ _Float16 sA [64 * 40];   // [row][k], stride 40 (bank-safe)
  __shared__ _Float16 sBt[64 * 40];   // [col][k]
  const int tid = threadIdx.x, lane = tid & 31, wave = tid >> 5;
  const int wm = wave >> 1, wn = wave & 1;
  const int mBase = blockIdx.y * 64, nBase = blockIdx.x * 64;
  const int lh = lane >> 4, ln = lane & 15;
  const int r4 = tid >> 2, c4 = (tid & 3) * 8;   // one b128 chunk per thread
  v8f acc0 = {}, acc1 = {};

  for (int kk = 0; kk < EMBED; kk += 32) {
    async_copy16(hA  + (size_t)(mBase + r4) * EMBED + kk + c4, sA  + r4 * 40 + c4);
    async_copy16(wBt + (size_t)(nBase + r4) * EMBED + kk + c4, sBt + r4 * 40 + c4);
    async_wait0();
    __syncthreads();
    const _Float16* ar = sA + (wm * 16 + ln) * 40 + lh * 16;
    v16h a = frag16(ar, ar + 8);
    const _Float16* bc0 = sBt + (wn * 32 + ln) * 40 + lh * 16;
    const _Float16* bc1 = sBt + (wn * 32 + 16 + ln) * 40 + lh * 16;
    v16h b0 = frag16(bc0, bc0 + 8);
    v16h b1 = frag16(bc1, bc1 + 8);
    acc0 = wmma_f16(a, b0, acc0);
    acc1 = wmma_f16(a, b1, acc1);
    __syncthreads();
  }

  // Epilogue: wave's 32-col tile is one head of one of {q,k,v}.
  // acc0 holds d in [0,16), acc1 d in [16,32): RoPE pairs are lane-local.
  const int c0 = nBase + wn * 32 + ln;
  const int sec = c0 >> 8;                 // 0=q 1=k 2=v
  const int head = (c0 & 255) >> 5;
  const int d0 = ln;
  const float invfreq = __expf(-(float)d0 * 0.5756462732485114f); // ln(1e4)/16
#pragma unroll
  for (int r = 0; r < 8; ++r) {
    int t = mBase + wm * 16 + r + lh * 8;
    int bb = t >> 10, n = t & (NSEQ - 1);
    float x0 = acc0[r], x1 = acc1[r], y0 = x0, y1 = x1;
    if (sec < 2) {
      float th = (float)n * invfreq;
      float sn = __sinf(th), cs = __cosf(th);
      y0 = x0 * cs - x1 * sn;
      y1 = x1 * cs + x0 * sn;
    }
    if (sec == 2) {           // V stored transposed: [b][h][d][n]
      size_t vb = ((size_t)bb * HEADS + head) * HDIM;
      vh[(vb + d0) * NSEQ + n]      = (_Float16)y0;
      vh[(vb + d0 + 16) * NSEQ + n] = (_Float16)y1;
    } else {                  // Q/K: [b][h][n][d]
      _Float16* dst = (sec == 0) ? qh : kh;
      size_t base = (((size_t)bb * HEADS + head) * NSEQ + n) * HDIM;
      dst[base + d0]      = (_Float16)y0;
      dst[base + d0 + 16] = (_Float16)y1;
    }
  }
}

// -------------------------- k4: fused flash attention + pairwise rel-bias MLP
// Grid (N/32, B). 256 threads = 8 waves = 8 heads; 32 queries per block.
__global__ void __launch_bounds__(256)
flash_attn_kernel(const _Float16* __restrict__ qh, const _Float16* __restrict__ kh,
                  const _Float16* __restrict__ vh, const _Float16* __restrict__ Ph,
                  const float* __restrict__ coords, const float* __restrict__ w1,
                  const float* __restrict__ b1, const float* __restrict__ w2,
                  const float* __restrict__ b2, _Float16* __restrict__ attn) {
  __shared__ float    sRelP[HEADS * 1024];    // 32KB rel[h][q][k]; P aliases it
  __shared__ _Float16 sVt[HEADS * 32 * 40];   // 20KB V^T tiles [h][d][k], pad 40
  __shared__ _Float16 sPi[32 * 72];           // queries' P (+b1), pad 72
  __shared__ _Float16 sPj[32 * 72];           // keys' P, pad 72
  __shared__ float    sW2[BH * HEADS];
  __shared__ float    sWd[BH];
  __shared__ float    sQC[64], sKC[64], sB2[HEADS];

  const int tid = threadIdx.x, lane = tid & 31, h = tid >> 5;
  const int lh = lane >> 4, ln = lane & 15;
  const int b = blockIdx.y, n0 = blockIdx.x * 32;
  const int tok0 = b * NSEQ + n0;

  // one-time fills
#pragma unroll
  for (int i = 0; i < 8; ++i) {
    int idx = tid + 256 * i, qq = idx >> 6, cc = idx & 63;
    sPi[qq * 72 + cc] = (_Float16)((float)Ph[(size_t)(tok0 + qq) * BH + cc] + b1[cc]);
  }
  if (tid < BH)         sWd[tid] = w1[2 * BH + tid];
  if (tid < BH * HEADS) sW2[tid] = w2[tid];
  if (tid < 64)         sQC[tid] = coords[(size_t)tok0 * 2 + tid];
  if (tid < HEADS)      sB2[tid] = b2[tid];

  // Q fragments for this wave's head
  const _Float16* qbase = qh + (((size_t)b * HEADS + h) * NSEQ + n0) * HDIM;
  const int ak0 = lh * 16;
  const _Float16* qr0 = qbase + ln * HDIM + ak0;
  const _Float16* qr1 = qbase + (16 + ln) * HDIM + ak0;
  v16h aq0 = frag16(qr0, qr0 + 8);
  v16h aq1 = frag16(qr1, qr1 + 8);

  float m0[8], m1[8], l0[8], l1[8];
#pragma unroll
  for (int r = 0; r < 8; ++r) { m0[r] = m1[r] = -1e30f; l0[r] = l1[r] = 0.f; }
  v8f o00 = {}, o01 = {}, o10 = {}, o11 = {};

  const float scale = 0.17677669529663687f;   // 1/sqrt(32)
  const _Float16* kbg = kh + ((size_t)b * HEADS + h) * NSEQ * HDIM;
  const _Float16* vtb = vh + ((size_t)b * HEADS + h) * HDIM * NSEQ; // [d][n]
  _Float16* ph = (_Float16*)&sRelP[h * 1024];   // own head's region (<=2.5KB/4KB)

  __syncthreads();

  for (int kt = 0; kt < 32; ++kt) {
    const int k0g = kt * 32;
    // keys' P rows: 4KB contiguous -> one async b128 chunk per thread
    async_copy16(Ph + ((size_t)(b * NSEQ + k0g)) * BH + tid * 8,
                 sPj + (tid >> 3) * 72 + (tid & 7) * 8);
    // own head's V^T tile: rows contiguous in global, 4 async chunks per lane
#pragma unroll
    for (int i = 0; i < 4; ++i) {
      int idx = lane + 32 * i;            // 128 chunks
      int d = idx >> 2, c4 = (idx & 3) * 8;
      async_copy16(vtb + (size_t)d * NSEQ + k0g + c4,
                   &sVt[h * 32 * 40 + d * 40 + c4]);
    }
    if (tid < 64) sKC[tid] = coords[(size_t)(b * NSEQ + k0g) * 2 + tid];
    if (kt + 1 < 32)
      __builtin_prefetch(kbg + (size_t)(k0g + 32) * HDIM, 0, 0);
    async_wait0();
    __syncthreads();

    // --- shared pairwise MLP: 1024 pairs, 4 per thread, all 8 heads at once --
#pragma unroll
    for (int pp = 0; pp < 4; ++pp) {
      int p = tid + 256 * pp, qq = p >> 5, kk = p & 31;
      float dx0 = sQC[qq * 2]     - sKC[kk * 2];
      float dx1 = sQC[qq * 2 + 1] - sKC[kk * 2 + 1];
      float dist = sqrtf(dx0 * dx0 + dx1 * dx1);
      float acc[HEADS];
#pragma unroll
      for (int hh = 0; hh < HEADS; ++hh) acc[hh] = sB2[hh];
      for (int c = 0; c < BH; ++c) {
        float t = (float)sPi[qq * 72 + c] - (float)sPj[kk * 72 + c]
                + dist * sWd[c];
        float gl = 0.5f * t * (1.0f + erff(t * 0.7071067811865475f));
#pragma unroll
        for (int hh = 0; hh < HEADS; ++hh)
          acc[hh] = fmaf(gl, sW2[c * HEADS + hh], acc[hh]);
      }
#pragma unroll
      for (int hh = 0; hh < HEADS; ++hh)
        sRelP[hh * 1024 + qq * 32 + kk] = acc[hh];
    }
    __syncthreads();

    // --- S = Q K^T : K fragments contiguous straight from global ---
    const _Float16* c0p = kbg + (size_t)(k0g + ln) * HDIM + lh * 16;
    const _Float16* c1p = kbg + (size_t)(k0g + 16 + ln) * HDIM + lh * 16;
    v16h bk0 = frag16(c0p, c0p + 8);
    v16h bk1 = frag16(c1p, c1p + 8);
    v8f z = {};
    v8f s00 = wmma_f16(aq0, bk0, z);
    v8f s01 = wmma_f16(aq0, bk1, z);
    v8f s10 = wmma_f16(aq1, bk0, z);
    v8f s11 = wmma_f16(aq1, bk1, z);

    const float* relh = &sRelP[h * 1024];
#pragma unroll
    for (int r = 0; r < 8; ++r) {
      int rq = r + lh * 8;
      s00[r] = s00[r] * scale + relh[rq * 32 + ln];
      s01[r] = s01[r] * scale + relh[rq * 32 + 16 + ln];
      s10[r] = s10[r] * scale + relh[(16 + rq) * 32 + ln];
      s11[r] = s11[r] * scale + relh[(16 + rq) * 32 + 16 + ln];
    }

    // --- online softmax (rows live across a 16-lane half); P -> LDS f16 ---
#pragma unroll
    for (int r = 0; r < 8; ++r) {
      int rq = r + lh * 8;
      { // qsub 0
        float pm = fmaxf(s00[r], s01[r]);
#pragma unroll
        for (int off = 8; off >= 1; off >>= 1) pm = fmaxf(pm, __shfl_xor(pm, off, 16));
        float mn = fmaxf(m0[r], pm);
        float corr = __expf(m0[r] - mn);
        float p0 = __expf(s00[r] - mn), p1 = __expf(s01[r] - mn);
        float ps = p0 + p1;
#pragma unroll
        for (int off = 8; off >= 1; off >>= 1) ps += __shfl_xor(ps, off, 16);
        l0[r] = l0[r] * corr + ps; m0[r] = mn;
        o00[r] *= corr; o01[r] *= corr;
        ph[rq * 40 + ln]      = (_Float16)p0;
        ph[rq * 40 + 16 + ln] = (_Float16)p1;
      }
      { // qsub 1
        float pm = fmaxf(s10[r], s11[r]);
#pragma unroll
        for (int off = 8; off >= 1; off >>= 1) pm = fmaxf(pm, __shfl_xor(pm, off, 16));
        float mn = fmaxf(m1[r], pm);
        float corr = __expf(m1[r] - mn);
        float p0 = __expf(s10[r] - mn), p1 = __expf(s11[r] - mn);
        float ps = p0 + p1;
#pragma unroll
        for (int off = 8; off >= 1; off >>= 1) ps += __shfl_xor(ps, off, 16);
        l1[r] = l1[r] * corr + ps; m1[r] = mn;
        o10[r] *= corr; o11[r] *= corr;
        ph[(16 + rq) * 40 + ln]      = (_Float16)p0;
        ph[(16 + rq) * 40 + 16 + ln] = (_Float16)p1;
      }
    }

    // --- O += P V : A-frags from LDS (b128), B-frags from V^T tile (b128) ---
    const _Float16* pr0 = ph + ln * 40 + ak0;
    const _Float16* pr1 = ph + (16 + ln) * 40 + ak0;
    v16h ap0 = frag16(pr0, pr0 + 8);
    v16h ap1 = frag16(pr1, pr1 + 8);
    const _Float16* vt = &sVt[h * 32 * 40];
    const int kb = lh * 16;
    const _Float16* vv0 = vt + ln * 40 + kb;
    const _Float16* vv1 = vt + (16 + ln) * 40 + kb;
    v16h bv0 = frag16(vv0, vv0 + 8);
    v16h bv1 = frag16(vv1, vv1 + 8);
    o00 = wmma_f16(ap0, bv0, o00);
    o01 = wmma_f16(ap0, bv1, o01);
    o10 = wmma_f16(ap1, bv0, o10);
    o11 = wmma_f16(ap1, bv1, o11);
    __syncthreads();
  }

  // writeback normalized output -> attn[b][n][h*32+d]
#pragma unroll
  for (int r = 0; r < 8; ++r) {
    int rq = r + lh * 8;
    float i0 = 1.0f / l0[r], i1 = 1.0f / l1[r];
    size_t ba = ((size_t)(b * NSEQ + n0 + rq)) * EMBED + h * HDIM;
    size_t bc = ((size_t)(b * NSEQ + n0 + 16 + rq)) * EMBED + h * HDIM;
    attn[ba + ln]      = (_Float16)(o00[r] * i0);
    attn[ba + 16 + ln] = (_Float16)(o01[r] * i0);
    attn[bc + ln]      = (_Float16)(o10[r] * i1);
    attn[bc + 16 + ln] = (_Float16)(o11[r] * i1);
  }
}

// ----------------------------------------------------- k5: output projection
__global__ void __launch_bounds__(256)
out_proj_kernel(const _Float16* __restrict__ A, const _Float16* __restrict__ Wt,
                const float* __restrict__ bo, float* __restrict__ out) {
  __shared__ _Float16 sA [64 * 40];
  __shared__ _Float16 sBt[64 * 40];
  const int tid = threadIdx.x, lane = tid & 31, wave = tid >> 5;
  const int wm = wave >> 1, wn = wave & 1;
  const int mBase = blockIdx.y * 64, nBase = blockIdx.x * 64;
  const int lh = lane >> 4, ln = lane & 15;
  const int r4 = tid >> 2, c4 = (tid & 3) * 8;
  v8f acc0 = {}, acc1 = {};
  for (int kk = 0; kk < EMBED; kk += 32) {
    async_copy16(A  + (size_t)(mBase + r4) * EMBED + kk + c4, sA  + r4 * 40 + c4);
    async_copy16(Wt + (size_t)(nBase + r4) * EMBED + kk + c4, sBt + r4 * 40 + c4);
    async_wait0();
    __syncthreads();
    const _Float16* ar = sA + (wm * 16 + ln) * 40 + lh * 16;
    v16h a = frag16(ar, ar + 8);
    const _Float16* bc0 = sBt + (wn * 32 + ln) * 40 + lh * 16;
    const _Float16* bc1 = sBt + (wn * 32 + 16 + ln) * 40 + lh * 16;
    v16h b0 = frag16(bc0, bc0 + 8);
    v16h b1 = frag16(bc1, bc1 + 8);
    acc0 = wmma_f16(a, b0, acc0);
    acc1 = wmma_f16(a, b1, acc1);
    __syncthreads();
  }
  const int c0 = nBase + wn * 32 + ln;
  const float bias0 = bo[c0], bias1 = bo[c0 + 16];
#pragma unroll
  for (int r = 0; r < 8; ++r) {
    int t = mBase + wm * 16 + r + lh * 8;
    out[(size_t)t * EMBED + c0]      = acc0[r] + bias0;
    out[(size_t)t * EMBED + c0 + 16] = acc1[r] + bias1;
  }
}

// ---------------------------------------------------------------------------
extern "C" void kernel_launch(void* const* d_in, const int* in_sizes, int n_in,
                              void* d_out, int out_size, void* d_ws, size_t ws_size,
                              hipStream_t stream) {
  (void)in_sizes; (void)n_in; (void)out_size; (void)ws_size;
  const float* x      = (const float*)d_in[0];
  const float* coords = (const float*)d_in[1];
  const float* vel    = (const float*)d_in[2];
  const float* ln_g   = (const float*)d_in[3];
  const float* ln_b   = (const float*)d_in[4];
  const float* w_qkv  = (const float*)d_in[5];
  const float* w_o    = (const float*)d_in[6];
  const float* b_o    = (const float*)d_in[7];
  const float* w1     = (const float*)d_in[8];
  const float* b1     = (const float*)d_in[9];
  const float* w2     = (const float*)d_in[10];
  const float* b2     = (const float*)d_in[11];
  float* out = (float*)d_out;

  char* ws = (char*)d_ws;
  _Float16* h_h     = (_Float16*)(ws + 0);          // 2048x256 f16
  _Float16* wqkv_t  = (_Float16*)(ws + 1048576);    // 768x256  f16 (W^T)
  _Float16* wo_t    = (_Float16*)(ws + 1441792);    // 256x256  f16 (W^T)
  _Float16* q_h     = (_Float16*)(ws + 1572864);    // [b][h][n][d]
  _Float16* k_h     = (_Float16*)(ws + 2621440);    // [b][h][n][d]
  _Float16* v_h     = (_Float16*)(ws + 3670016);    // [b][h][d][n]  (transposed)
  _Float16* P_h     = (_Float16*)(ws + 4718592);    // 2048x64  f16
  _Float16* attn_h  = (_Float16*)(ws + 4980736);    // 2048x256 f16

  convert_weights_kernel<<<256, 256, 0, stream>>>(w_qkv, w_o, wqkv_t, wo_t);
  layernorm_kernel<<<NTOK / 8, 256, 0, stream>>>(x, ln_g, ln_b, h_h);
  token_feat_kernel<<<NTOK * BH / 256, 256, 0, stream>>>(coords, vel, w1, P_h);
  qkv_gemm_rope_kernel<<<dim3(12, 32), 256, 0, stream>>>(h_h, wqkv_t, q_h, k_h, v_h);
  flash_attn_kernel<<<dim3(NSEQ / 32, 2), 256, 0, stream>>>(
      q_h, k_h, v_h, P_h, coords, w1, b1, w2, b2, attn_h);
  out_proj_kernel<<<dim3(4, 32), 256, 0, stream>>>(attn_h, wo_t, b_o, out);
}
